// Block_53618371724026
// MI455X (gfx1250) — compile-verified, matrix operand order checked by reading, not run on hardware
//
#include <hip/hip_runtime.h>

// ---------------------------------------------------------------------------
// CDNA5 (gfx1250) transformer block: pre-LN attention + FFN, bf16 WMMA GEMMs.
// A tiles staged by the Tensor Data Mover (TENSORcnt), scores tiles by async
// LDS DMA (ASYNCcnt); double-buffered LDS; wave32.
// ---------------------------------------------------------------------------

typedef __attribute__((ext_vector_type(16))) __bf16          v16bf;
typedef __attribute__((ext_vector_type(8)))  float           v8f;
typedef __attribute__((ext_vector_type(8)))  unsigned short  ushort8;
typedef __attribute__((ext_vector_type(4)))  unsigned int    uint4v;
typedef __attribute__((ext_vector_type(8)))  int             int8v;
typedef __attribute__((ext_vector_type(4)))  int             int4v;

union FragCast { ushort8 u[2]; v16bf v; };

#define BM  128
#define BN  128
#define BK  32
#define BKP 40           // padded LDS row stride (elements): 64B row + 16B pad

#if defined(__has_builtin)
#if __has_builtin(__builtin_amdgcn_tensor_load_to_lds)
#define HAVE_TDM 1
#endif
#endif
#ifndef HAVE_TDM
#define HAVE_TDM 0
#endif

__device__ __forceinline__ unsigned short f2bf(float f) {
  unsigned int u = __float_as_uint(f);
  u += 0x7FFFu + ((u >> 16) & 1u);     // round-to-nearest-even
  return (unsigned short)(u >> 16);
}

// Async DMA of 16 bytes global -> LDS (tracked by ASYNCcnt).
__device__ __forceinline__ void async_g2l_b128(unsigned lds_addr, const void* gaddr) {
  asm volatile("global_load_async_to_lds_b128 %0, %1, off"
               :: "v"(lds_addr), "v"(gaddr) : "memory");
}
__device__ __forceinline__ void wait_async0() {
  asm volatile("s_wait_asynccnt 0x0" ::: "memory");
}
__device__ __forceinline__ void wait_tensor0() {
#if defined(__has_builtin) && __has_builtin(__builtin_amdgcn_s_wait_tensorcnt)
  __builtin_amdgcn_s_wait_tensorcnt(0);
#else
  asm volatile("s_wait_tensorcnt 0x0" ::: "memory");
#endif
}

#if HAVE_TDM
// One TDM descriptor: DMA a [tile_d1 x tile_d0] bf16 tile (row stride
// `stride0` elements) from global into LDS at `lds_addr`, inserting 16B of
// LDS padding after every 64B row chunk (-> BKP=40 element rows for BK=32).
__device__ __forceinline__ void tdm_load_tile_bf16(unsigned lds_addr,
                                                   unsigned long long gaddr,
                                                   unsigned tile_d0, unsigned tile_d1,
                                                   unsigned long long stride0)
{
  const unsigned TD = 1u << 20;            // tensor dims large: no OOB clipping
  uint4v g0;
  g0[0] = 1u;                              // count=1, user descriptor
  g0[1] = lds_addr;                        // LDS byte address
  g0[2] = (unsigned)(gaddr & 0xFFFFFFFFull);
  g0[3] = (unsigned)((gaddr >> 32) & 0x1FFFFFFull) | (2u << 30);  // type=2 (image)

  int8v g1;
  g1[0] = (int)(0x10000u                   // data_size = 1 (2 bytes)
              | 0x100000u                  // pad_enable
              | (3u << 22)                 // pad_interval: 16 DWORDs (64B)
              | (3u << 25));               // pad_amount:   4 DWORDs (16B)
  g1[1] = (int)((TD & 0xFFFFu) << 16);                       // tensor_dim0 lo16
  g1[2] = (int)((TD >> 16) | ((TD & 0xFFFFu) << 16));        // dim0 hi | dim1 lo
  g1[3] = (int)((TD >> 16) | (tile_d0 << 16));               // dim1 hi | tile_dim0
  g1[4] = (int)tile_d1;                                      // tile_dim1 | tile_dim2=0
  g1[5] = (int)(unsigned)(stride0 & 0xFFFFFFFFull);          // dim0_stride lo32
  g1[6] = (int)(unsigned)((stride0 >> 32) & 0xFFFFull);      // dim0_stride hi16
  g1[7] = 0;

  int4v z4 = {0, 0, 0, 0};
#if __clang_major__ >= 23
  int8v z8 = {0, 0, 0, 0, 0, 0, 0, 0};
  __builtin_amdgcn_tensor_load_to_lds(g0, g1, z4, z4, z8, 0);
#else
  __builtin_amdgcn_tensor_load_to_lds(g0, g1, z4, z4, 0);
#endif
}
#endif

// 16-bit A/B fragment of a 16x16x32 WMMA from an LDS row-major [16][BKP] tile.
// Lanes 0-15: row r=lane, K chunks {0..7,16..23}; lanes 16-31: row r=lane-16,
// K chunks {8..15,24..31}.  Two ds_load_b128 per fragment.
__device__ __forceinline__ v16bf frag_from_lds(const unsigned short* base, int lane) {
  int r  = lane & 15;
  int kh = (lane >> 4) << 3;           // 0 or 8
  const unsigned short* p = base + r * BKP;
  FragCast f;
  f.u[0] = *(const ushort8*)(p + kh);
  f.u[1] = *(const ushort8*)(p + 16 + kh);
  return f.v;
}

// ---------------------------------------------------------------------------
// Generic batched bf16 GEMM:  C[z] = A[z] * B[z] (+bias)(relu)(+residual)
// z1 = z / batch_div, z0 = z % batch_div select batch offsets.
// A tile staged by TDM (one tensor_load_to_lds per workgroup per K-step);
// B ([K x N] row-major) staged transposed (Bs[n][k]) via registers so B
// fragments load with the same pattern as A.
// ---------------------------------------------------------------------------
__global__ void __launch_bounds__(256)
gemm_bf16_wmma(const unsigned short* __restrict__ A,  long long sAz1, long long sAz0, int lda,
               const unsigned short* __restrict__ Bm, long long sBz1, long long sBz0, int ldb,
               const float* __restrict__ bias,
               const float* __restrict__ resid, int ldr,
               float* __restrict__ Cf, unsigned short* __restrict__ Cb,
               long long sCz1, long long sCz0, int ldc,
               int K, int batch_div, int relu)
{
  __shared__ unsigned short As[2][BM * BKP];
  __shared__ unsigned short Bs[2][BN * BKP];

  const int z  = blockIdx.z;
  const int z1 = z / batch_div;
  const int z0 = z - z1 * batch_div;
  A  += (size_t)(z1 * sAz1 + z0 * sAz0);
  Bm += (size_t)(z1 * sBz1 + z0 * sBz0);
  const long long co = z1 * sCz1 + z0 * sCz0;

  const int m0   = blockIdx.y * BM;
  const int n0   = blockIdx.x * BN;
  const int tid  = threadIdx.x;
  const int lane = tid & 31;
  const int wid  = tid >> 5;
  const int wm   = (wid >> 1) * 32;
  const int wn   = (wid & 1)  * 64;

  // B staging coordinates (2 chunks per thread, fully unrolled).
  const int bk  = tid >> 4;                // B k rows: bk and bk+16
  const int bnc = (tid & 15) << 3;         // B N offset (8-elem chunk)
  const unsigned short* pB0 = Bm + (size_t)bk * ldb + n0 + bnc;
  const unsigned short* pB1 = pB0 + (size_t)16 * ldb;
  const size_t stepB = (size_t)BK * ldb;

  const unsigned bufBytes = (unsigned)(BM * BKP * sizeof(unsigned short));
  const unsigned ldsA     = (unsigned)(size_t)&As[0][0];

#if HAVE_TDM
  const unsigned long long gA =
      (unsigned long long)(size_t)(const void*)(A + (size_t)m0 * lda);
#else
  const int arow = tid >> 2;
  const int akc  = (tid & 3) << 3;
  const unsigned short* pA0 = A + (size_t)(m0 + arow) * lda + akc;
  const unsigned short* pA1 = pA0 + (size_t)64 * lda;
  const unsigned lA0 = ldsA + (unsigned)((arow * BKP + akc) * 2);
  const unsigned lA1 = ldsA + (unsigned)(((arow + 64) * BKP + akc) * 2);
#endif

  v8f acc[2][4];
#pragma unroll
  for (int i = 0; i < 2; ++i)
#pragma unroll
    for (int j = 0; j < 4; ++j)
      acc[i][j] = (v8f){0.f,0.f,0.f,0.f,0.f,0.f,0.f,0.f};

  // ---- prologue: prefetch tile 0 into buffer 0 ----
#if HAVE_TDM
  if (wid == 0) tdm_load_tile_bf16(ldsA, gA, BK, BM, (unsigned long long)lda);
#else
  async_g2l_b128(lA0, pA0);
  async_g2l_b128(lA1, pA1);
#endif
  {
    ushort8 r0 = *(const ushort8*)pB0;
    ushort8 r1 = *(const ushort8*)pB1;
    unsigned short* d0 = &Bs[0][bnc * BKP + bk];
#pragma unroll
    for (int j = 0; j < 8; ++j) { d0[j * BKP] = r0[j]; d0[j * BKP + 16] = r1[j]; }
  }
  const unsigned short* nB0 = pB0 + stepB;   // next-tile B pointers (incremental)
  const unsigned short* nB1 = pB1 + stepB;

  const int ntiles = K / BK;
  for (int it = 0; it < ntiles; ++it) {
    const int cur = it & 1;
    const int nxt = cur ^ 1;
#if HAVE_TDM
    if (wid == 0) wait_tensor0();  // tile `it` committed to LDS by the TDM
#else
    wait_async0();
#endif
    __syncthreads();               // all waves' stores/DMAs for tile `it` visible

    const bool more = (it + 1) < ntiles;
    ushort8 r0, r1;
    if (more) {
#if HAVE_TDM
      if (wid == 0)
        tdm_load_tile_bf16(ldsA + nxt * bufBytes,
                           gA + (unsigned long long)(it + 1) * (BK * 2),
                           BK, BM, (unsigned long long)lda);
#else
      const int ko = (it + 1) * BK;
      async_g2l_b128(lA0 + nxt * bufBytes, pA0 + ko);
      async_g2l_b128(lA1 + nxt * bufBytes, pA1 + ko);
#endif
      r0 = *(const ushort8*)nB0; nB0 += stepB;
      r1 = *(const ushort8*)nB1; nB1 += stepB;
    }

    v16bf af[2], bfr[4];
#pragma unroll
    for (int i = 0; i < 2; ++i) af[i]  = frag_from_lds(&As[cur][(wm + i * 16) * BKP], lane);
#pragma unroll
    for (int j = 0; j < 4; ++j) bfr[j] = frag_from_lds(&Bs[cur][(wn + j * 16) * BKP], lane);

    if (more) {                    // transpose-scatter next B tile behind the loads
      unsigned short* d0 = &Bs[nxt][bnc * BKP + bk];
#pragma unroll
      for (int j = 0; j < 8; ++j) { d0[j * BKP] = r0[j]; d0[j * BKP + 16] = r1[j]; }
    }

#pragma unroll
    for (int i = 0; i < 2; ++i)
#pragma unroll
      for (int j = 0; j < 4; ++j)
        acc[i][j] = __builtin_amdgcn_wmma_f32_16x16x32_bf16(
            false, af[i], false, bfr[j], (short)0, acc[i][j], false, false);
  }

  // epilogue: VGPR r, lanes 0-15: (M=r, N=lane); lanes 16-31: (M=8+r, N=lane-16)
  const int cn = lane & 15;
  const int mh = (lane >> 4) << 3;
#pragma unroll
  for (int i = 0; i < 2; ++i) {
#pragma unroll
    for (int j = 0; j < 4; ++j) {
      const int rb = m0 + wm + i * 16 + mh;
      const int cb = n0 + wn + j * 16 + cn;
      const float bv = bias ? bias[cb] : 0.f;
#pragma unroll
      for (int r = 0; r < 8; ++r) {
        const int gm = rb + r;
        float v = acc[i][j][r] + bv;
        if (relu)  v = fmaxf(v, 0.f);
        if (resid) v += resid[(size_t)gm * ldr + cb];
        const long long idx = co + (long long)gm * ldc + cb;
        if (Cf) Cf[idx] = v;
        if (Cb) Cb[idx] = f2bf(v);
      }
    }
  }
}

// ---------------------------------------------------------------------------
// Attention scores: S[z] = scale * Q[z] (T x HD) * K[z]^T  (z = h*B + b).
// Both operands row-major [T][HD] -> both tiles staged entirely by async DMA.
// Tiles strictly above the causal diagonal are skipped.
// ---------------------------------------------------------------------------
__global__ void __launch_bounds__(256)
attn_scores_wmma(const unsigned short* __restrict__ Q,
                 const unsigned short* __restrict__ Kt,
                 float* __restrict__ S, float scale)
{
  const int T = 512, HD = 512;
  __shared__ unsigned short Qs[2][BM * BKP];
  __shared__ unsigned short Ks[2][BN * BKP];

  const int m0 = blockIdx.y * BM;
  const int n0 = blockIdx.x * BN;
  if (n0 > m0 + (BM - 1)) return;          // fully-masked tile (uniform branch)

  const int z = blockIdx.z;
  const unsigned short* Qz = Q  + (size_t)z * T * HD;
  const unsigned short* Kz = Kt + (size_t)z * T * HD;
  float* Sz = S + (size_t)z * T * T;

  const int tid  = threadIdx.x;
  const int lane = tid & 31;
  const int wid  = tid >> 5;
  const int wm   = (wid >> 1) * 32;
  const int wn   = (wid & 1)  * 64;

  const int row = tid >> 2;                // rows: row and row+64
  const int kc  = (tid & 3) << 3;

  const unsigned short* pQ0 = Qz + (size_t)(m0 + row) * HD + kc;
  const unsigned short* pQ1 = pQ0 + (size_t)64 * HD;
  const unsigned short* pK0 = Kz + (size_t)(n0 + row) * HD + kc;
  const unsigned short* pK1 = pK0 + (size_t)64 * HD;

  const unsigned bufBytes = (unsigned)(BM * BKP * sizeof(unsigned short));
  const unsigned lQ0 = (unsigned)(size_t)&Qs[0][row * BKP + kc];
  const unsigned lQ1 = (unsigned)(size_t)&Qs[0][(row + 64) * BKP + kc];
  const unsigned lK0 = (unsigned)(size_t)&Ks[0][row * BKP + kc];
  const unsigned lK1 = (unsigned)(size_t)&Ks[0][(row + 64) * BKP + kc];

  v8f acc[2][4];
#pragma unroll
  for (int i = 0; i < 2; ++i)
#pragma unroll
    for (int j = 0; j < 4; ++j)
      acc[i][j] = (v8f){0.f,0.f,0.f,0.f,0.f,0.f,0.f,0.f};

  // prologue: tile 0 -> buffer 0 (all async)
  async_g2l_b128(lQ0, pQ0);
  async_g2l_b128(lQ1, pQ1);
  async_g2l_b128(lK0, pK0);
  async_g2l_b128(lK1, pK1);

  const int ntiles = HD / BK;
  for (int it = 0; it < ntiles; ++it) {
    const int cur = it & 1;
    const int nxt = cur ^ 1;
    wait_async0();
    __syncthreads();

    if (it + 1 < ntiles) {
      const int ko = (it + 1) * BK;
      async_g2l_b128(lQ0 + nxt * bufBytes, pQ0 + ko);
      async_g2l_b128(lQ1 + nxt * bufBytes, pQ1 + ko);
      async_g2l_b128(lK0 + nxt * bufBytes, pK0 + ko);
      async_g2l_b128(lK1 + nxt * bufBytes, pK1 + ko);
    }

    v16bf af[2], bfr[4];
#pragma unroll
    for (int i = 0; i < 2; ++i) af[i]  = frag_from_lds(&Qs[cur][(wm + i * 16) * BKP], lane);
#pragma unroll
    for (int j = 0; j < 4; ++j) bfr[j] = frag_from_lds(&Ks[cur][(wn + j * 16) * BKP], lane);

#pragma unroll
    for (int i = 0; i < 2; ++i)
#pragma unroll
      for (int j = 0; j < 4; ++j)
        acc[i][j] = __builtin_amdgcn_wmma_f32_16x16x32_bf16(
            false, af[i], false, bfr[j], (short)0, acc[i][j], false, false);
  }

  const int cn = lane & 15;
  const int mh = (lane >> 4) << 3;
#pragma unroll
  for (int i = 0; i < 2; ++i)
#pragma unroll
    for (int j = 0; j < 4; ++j) {
      const int rb = m0 + wm + i * 16 + mh;
      const int cb = n0 + wn + j * 16 + cn;
#pragma unroll
      for (int r = 0; r < 8; ++r)
        Sz[(size_t)(rb + r) * T + cb] = acc[i][j][r] * scale;
    }
}

// ---------------------------------------------------------------------------
// LayerNorm over D=2048, output bf16 (one block per row)
// ---------------------------------------------------------------------------
__global__ void __launch_bounds__(256)
layernorm_bf16(const float* __restrict__ x, const float* __restrict__ g,
               const float* __restrict__ be, unsigned short* __restrict__ out)
{
  const int D = 2048;
  const int row = blockIdx.x;
  const float* xr = x + (size_t)row * D;
  __shared__ float s1[256], s2[256];
  const int tid = threadIdx.x;
  float sum = 0.f, sq = 0.f;
  for (int i = tid; i < D; i += 256) { float v = xr[i]; sum += v; sq += v * v; }
  s1[tid] = sum; s2[tid] = sq; __syncthreads();
  for (int o = 128; o > 0; o >>= 1) {
    if (tid < o) { s1[tid] += s1[tid + o]; s2[tid] += s2[tid + o]; }
    __syncthreads();
  }
  const float mu  = s1[0] * (1.f / D);
  const float var = s2[0] * (1.f / D) - mu * mu;
  const float rs  = rsqrtf(var + 1e-5f);
  unsigned short* orow = out + (size_t)row * D;
  for (int i = tid; i < D; i += 256)
    orow[i] = f2bf((xr[i] - mu) * rs * g[i] + be[i]);
}

// ---------------------------------------------------------------------------
// Causal softmax over rows of S [H*B*T][T], bf16 probabilities out (0 masked)
// ---------------------------------------------------------------------------
__global__ void __launch_bounds__(256)
softmax_causal(const float* __restrict__ S, unsigned short* __restrict__ P)
{
  const int T = 512;
  const int rid = blockIdx.x;
  const int t = rid & (T - 1);
  const float* sr = S + (size_t)rid * T;
  unsigned short* pr = P + (size_t)rid * T;
  __shared__ float red[256];
  const int tid = threadIdx.x;

  const float v0 = (tid       <= t) ? sr[tid]       : -3.4e38f;
  const float v1 = (tid + 256 <= t) ? sr[tid + 256] : -3.4e38f;
  red[tid] = fmaxf(v0, v1); __syncthreads();
  for (int o = 128; o > 0; o >>= 1) {
    if (tid < o) red[tid] = fmaxf(red[tid], red[tid + o]);
    __syncthreads();
  }
  const float m = red[0]; __syncthreads();

  const float e0 = (tid       <= t) ? __expf(v0 - m) : 0.f;
  const float e1 = (tid + 256 <= t) ? __expf(v1 - m) : 0.f;
  red[tid] = e0 + e1; __syncthreads();
  for (int o = 128; o > 0; o >>= 1) {
    if (tid < o) red[tid] += red[tid + o];
    __syncthreads();
  }
  const float inv = 1.f / red[0];
  pr[tid]       = f2bf(e0 * inv);
  pr[tid + 256] = f2bf(e1 * inv);
}

// ---------------------------------------------------------------------------
// f32 -> bf16 bulk conversion (grid-stride)
// ---------------------------------------------------------------------------
__global__ void __launch_bounds__(256)
cvt_f32_bf16(const float* __restrict__ in, unsigned short* __restrict__ out, long long n)
{
  long long i = (long long)blockIdx.x * 256 + threadIdx.x;
  const long long stride = (long long)gridDim.x * 256;
  for (; i < n; i += stride) out[i] = f2bf(in[i]);
}

// ---------------------------------------------------------------------------
// Host-side orchestration
// ---------------------------------------------------------------------------
extern "C" void kernel_launch(void* const* d_in, const int* in_sizes, int n_in,
                              void* d_out, int out_size, void* d_ws, size_t ws_size,
                              hipStream_t stream) {
  (void)in_sizes; (void)n_in; (void)out_size; (void)ws_size;

  const int Bb = 16, T = 512, D = 2048, H = 4, HD = 512;
  const int BT = Bb * T;        // 8192 rows
  const int DF = 4 * D;         // 8192 FFN dim

  const float* x  = (const float*)d_in[0];
  const float* Wq = (const float*)d_in[1];
  const float* Wk = (const float*)d_in[2];
  const float* Wv = (const float*)d_in[3];
  const float* Wo = (const float*)d_in[4];
  const float* bo = (const float*)d_in[5];
  const float* W1 = (const float*)d_in[6];
  const float* b1 = (const float*)d_in[7];
  const float* W2 = (const float*)d_in[8];
  const float* b2 = (const float*)d_in[9];
  const float* g1  = (const float*)d_in[10];
  const float* be1 = (const float*)d_in[11];
  const float* g2  = (const float*)d_in[12];
  const float* be2 = (const float*)d_in[13];
  float* out = (float*)d_out;

  // workspace bump allocator (256B aligned)
  char* wsp = (char*)d_ws;
  auto alloc = [&](size_t bytes) -> void* {
    void* p = wsp; wsp += (bytes + 255) & ~(size_t)255; return p;
  };
  unsigned short* hbf   = (unsigned short*)alloc((size_t)BT * D  * 2);  // LN1(x) bf16
  unsigned short* wqb   = (unsigned short*)alloc((size_t)H * D * HD * 2);
  unsigned short* wkb   = (unsigned short*)alloc((size_t)H * D * HD * 2);
  unsigned short* wvb   = (unsigned short*)alloc((size_t)H * D * HD * 2);
  unsigned short* wob   = (unsigned short*)alloc((size_t)D * D  * 2);
  unsigned short* w1b   = (unsigned short*)alloc((size_t)D * DF * 2);
  unsigned short* w2b   = (unsigned short*)alloc((size_t)DF * D * 2);
  unsigned short* qbf   = (unsigned short*)alloc((size_t)H * BT * HD * 2); // [H][B*T][HD]
  unsigned short* kbf   = (unsigned short*)alloc((size_t)H * BT * HD * 2);
  unsigned short* vbf   = (unsigned short*)alloc((size_t)H * BT * HD * 2);
  float*          Sf    = (float*)         alloc((size_t)H * Bb * T * T * 4);
  unsigned short* pbf   = (unsigned short*)alloc((size_t)H * Bb * T * T * 2);
  unsigned short* ctxbf = (unsigned short*)alloc((size_t)BT * D  * 2);  // heads concat
  float*          x2    = (float*)         alloc((size_t)BT * D  * 4);
  unsigned short* h2bf  = (unsigned short*)alloc((size_t)BT * D  * 2);
  unsigned short* ubf   = (unsigned short*)alloc((size_t)BT * DF * 2);  // relu(h2 W1 + b1)

  // --- weight conversions to bf16 ---
  cvt_f32_bf16<<<2048, 256, 0, stream>>>(Wq, wqb, (long long)H * D * HD);
  cvt_f32_bf16<<<2048, 256, 0, stream>>>(Wk, wkb, (long long)H * D * HD);
  cvt_f32_bf16<<<2048, 256, 0, stream>>>(Wv, wvb, (long long)H * D * HD);
  cvt_f32_bf16<<<2048, 256, 0, stream>>>(Wo, wob, (long long)D * D);
  cvt_f32_bf16<<<2048, 256, 0, stream>>>(W1, w1b, (long long)D * DF);
  cvt_f32_bf16<<<2048, 256, 0, stream>>>(W2, w2b, (long long)DF * D);

  // --- LN1 ---
  layernorm_bf16<<<BT, 256, 0, stream>>>(x, g1, be1, hbf);

  // --- Q/K/V projections: batch over heads (z1 = h), M=8192 N=512 K=2048 ---
  const dim3 gqkv(HD / BN, BT / BM, H);
  gemm_bf16_wmma<<<gqkv, 256, 0, stream>>>(
      hbf, 0, 0, D,   wqb, (long long)D * HD, 0, HD,
      nullptr, nullptr, 0,
      nullptr, qbf, (long long)BT * HD, 0, HD,  D, 1, 0);
  gemm_bf16_wmma<<<gqkv, 256, 0, stream>>>(
      hbf, 0, 0, D,   wkb, (long long)D * HD, 0, HD,
      nullptr, nullptr, 0,
      nullptr, kbf, (long long)BT * HD, 0, HD,  D, 1, 0);
  gemm_bf16_wmma<<<gqkv, 256, 0, stream>>>(
      hbf, 0, 0, D,   wvb, (long long)D * HD, 0, HD,
      nullptr, nullptr, 0,
      nullptr, vbf, (long long)BT * HD, 0, HD,  D, 1, 0);

  // --- scores S = scale * Q K^T, z = h*B + b, causal upper tiles skipped ---
  const float scale = 1.f / sqrtf((float)HD);
  attn_scores_wmma<<<dim3(T / BN, T / BM, H * Bb), 256, 0, stream>>>(qbf, kbf, Sf, scale);

  // --- causal softmax -> bf16 probabilities ---
  softmax_causal<<<H * Bb * T, 256, 0, stream>>>(Sf, pbf);

  // --- ctx = P @ V, batch z = h*B + b (batch_div=B): out into [B][T][h*HD..] ---
  gemm_bf16_wmma<<<dim3(HD / BN, T / BM, H * Bb), 256, 0, stream>>>(
      pbf, (long long)Bb * T * T,  (long long)T * T,  T,
      vbf, (long long)Bb * T * HD, (long long)T * HD, HD,
      nullptr, nullptr, 0,
      nullptr, ctxbf, (long long)HD, (long long)T * D, D,  T, Bb, 0);

  // --- x2 = x + ctx @ Wo + bo ---
  gemm_bf16_wmma<<<dim3(D / BN, BT / BM, 1), 256, 0, stream>>>(
      ctxbf, 0, 0, D,   wob, 0, 0, D,
      bo, x, D,
      x2, nullptr, 0, 0, D,  D, 1, 0);

  // --- LN2 ---
  layernorm_bf16<<<BT, 256, 0, stream>>>(x2, g2, be2, h2bf);

  // --- FFN1: U = relu(h2 @ W1 + b1), bf16 out ---
  gemm_bf16_wmma<<<dim3(DF / BN, BT / BM, 1), 256, 0, stream>>>(
      h2bf, 0, 0, D,   w1b, 0, 0, DF,
      b1, nullptr, 0,
      nullptr, ubf, 0, 0, DF,  D, 1, 1);

  // --- FFN2: out = x2 + U @ W2 + b2 ---
  gemm_bf16_wmma<<<dim3(D / BN, BT / BM, 1), 256, 0, stream>>>(
      ubf, 0, 0, DF,   w2b, 0, 0, D,
      b2, x2, D,
      out, nullptr, 0, 0, D,  DF, 1, 0);
}
